// RNNGraphConvModule_45079976739288
// MI455X (gfx1250) — matrix operand order repaired and checked by reading, not compiled, not run on hardware
//
#include <hip/hip_runtime.h>
#include <hip/hip_bf16.h>

#define NC    16
#define FDIM  13
#define FHID  64

typedef __attribute__((ext_vector_type(16))) _Float16 v16h;
typedef __attribute__((ext_vector_type(8)))  _Float16 v8h;
typedef __attribute__((ext_vector_type(8)))  float    v8f;

#if __has_builtin(__builtin_amdgcn_tensor_load_to_lds)
#define USE_TDM 1
typedef __attribute__((ext_vector_type(4))) unsigned int u32x4;
typedef __attribute__((ext_vector_type(8))) int          i32x8;
typedef __attribute__((ext_vector_type(4))) int          i32x4;
#else
#define USE_TDM 0
#endif

// ---------- WMMA lane/VGPR layouts (CDNA5 ISA 7.12.2, wave32) ----------
// f16 A 16x32 (MxK): lane L (m=L%16, hi=L/16) holds elements i=0..7 -> K=hi*8+i,
//                    i=8..15 -> K=16+hi*8+(i-8). Two contiguous 8-half runs.
// f16 B 32x16 (KxN): lane L (n=L%16, hi=L/16) holds elements i=0..7 -> K=hi*8+i,
//                    i=8..15 -> K=16+hi*8+(i-8). Two 8-half runs (N-major storage).
// f32 C/D 16x16:     lane L, reg v -> row m=v+hi*8, col n=L%16.
__device__ __forceinline__ v16h cat8(v8h lo, v8h hi) {
  return __builtin_shufflevector(lo, hi, 0, 1, 2, 3, 4, 5, 6, 7,
                                 8, 9, 10, 11, 12, 13, 14, 15);
}
__device__ __forceinline__ float bf16bits_to_f32(unsigned short u) {
  union { unsigned int i; float f; } c; c.i = ((unsigned int)u) << 16; return c.f;
}
__device__ __forceinline__ float sigmoidf_(float x) { return 1.0f / (1.0f + __expf(-x)); }
__device__ __forceinline__ float relu_(float x) { return x > 0.0f ? x : 0.0f; }

// =====================================================================
// Kernel 1: filter net.  W_bf16[e,16,16] = relu(ef@w1+b1)@w2 + b2
// Block = 256 thr = 8 waves; wave owns a 16-edge M-tile (128 edges/block).
// All WMMA operands are ds_load_b128 pairs (weights stored N-major in LDS).
// =====================================================================
__global__ void fnet_wmma_kernel(const float* __restrict__ ef,
                                 const float* __restrict__ w1,
                                 const float* __restrict__ b1,
                                 const float* __restrict__ w2,
                                 const float* __restrict__ b2,
                                 __hip_bfloat16* __restrict__ Wout,
                                 int E) {
  __shared__ __align__(16) _Float16 efs[128 * 32];   // [m][k] K padded 13->32
  __shared__ __align__(16) _Float16 w1t[64 * 32];    // [j][k] transposed, padded
  __shared__ __align__(16) _Float16 w2t[256 * 64];   // [j][k] transposed
  __shared__ float b1s[64];
  __shared__ float b2s[256];
  __shared__ __align__(16) _Float16 hid[8][16 * 64]; // per-wave relu hidden [m][k]

  const int tid  = threadIdx.x;
  const int wave = tid >> 5;
  const int lane = tid & 31;
  const int hi   = lane >> 4;
  const int mn   = lane & 15;
  const int e0blk = blockIdx.x * 128;

  for (int i = tid; i < 128 * 32; i += 256) {
    int m = i >> 5, k = i & 31;
    int e = min(e0blk + m, E - 1);
    efs[i] = (k < FDIM) ? (_Float16)ef[(long)e * FDIM + k] : (_Float16)0.0f;
  }
  for (int i = tid; i < 64 * 32; i += 256) {
    int j = i >> 5, k = i & 31;
    w1t[i] = (k < FDIM) ? (_Float16)w1[k * 64 + j] : (_Float16)0.0f;
  }
  for (int i = tid; i < 64; i += 256) b1s[i] = b1[i];
  for (int i = tid; i < 64 * 256; i += 256) {      // coalesced read, transposed write
    int k = i >> 8, j = i & 255;
    w2t[j * 64 + k] = (_Float16)w2[i];
  }
  for (int i = tid; i < 256; i += 256) b2s[i] = b2[i];
  __syncthreads();

  const int e0 = e0blk + wave * 16;

  // A: edge features (2 x b128)
  const _Float16* ar = &efs[(wave * 16 + mn) * 32];
  v16h a = cat8(*(const v8h*)(ar + hi * 8), *(const v8h*)(ar + 16 + hi * 8));

  // GEMM1: hidden = relu(A @ w1 + b1)
#pragma unroll
  for (int jn = 0; jn < 4; ++jn) {
    const _Float16* br = &w1t[(jn * 16 + mn) * 32];
    v16h b = cat8(*(const v8h*)(br + hi * 8), *(const v8h*)(br + 16 + hi * 8));
    v8f c;
    float bias = b1s[jn * 16 + mn];
#pragma unroll
    for (int v = 0; v < 8; ++v) c[v] = bias;
    c = __builtin_amdgcn_wmma_f32_16x16x32_f16(false, a, false, b, (short)0, c, false, false);
#pragma unroll
    for (int v = 0; v < 8; ++v)
      hid[wave][(v + hi * 8) * 64 + jn * 16 + mn] = (_Float16)relu_(c[v]);
  }
  __syncthreads();

  // GEMM2: W = hidden @ w2 + b2  (K=64 -> two K32 WMMAs per 16-col tile)
  const _Float16* hr = &hid[wave][mn * 64];
  v16h a0 = cat8(*(const v8h*)(hr + hi * 8),      *(const v8h*)(hr + 16 + hi * 8));
  v16h a1 = cat8(*(const v8h*)(hr + 32 + hi * 8), *(const v8h*)(hr + 48 + hi * 8));
#pragma unroll 4
  for (int jn = 0; jn < 16; ++jn) {
    const _Float16* br = &w2t[(jn * 16 + mn) * 64];
    v16h b0  = cat8(*(const v8h*)(br + hi * 8),      *(const v8h*)(br + 16 + hi * 8));
    v16h b1v = cat8(*(const v8h*)(br + 32 + hi * 8), *(const v8h*)(br + 48 + hi * 8));
    v8f c;
    float bias = b2s[jn * 16 + mn];
#pragma unroll
    for (int v = 0; v < 8; ++v) c[v] = bias;
    c = __builtin_amdgcn_wmma_f32_16x16x32_f16(false, a0, false, b0,  (short)0, c, false, false);
    c = __builtin_amdgcn_wmma_f32_16x16x32_f16(false, a1, false, b1v, (short)0, c, false, false);
#pragma unroll
    for (int v = 0; v < 8; ++v) {
      int e = e0 + v + hi * 8;
      if (e < E) Wout[(long)e * 256 + jn * 16 + mn] = __float2bfloat16(c[v]);
    }
  }
}

// =====================================================================
// Degree / inverse degree
// =====================================================================
__global__ void deg_kernel(const int* __restrict__ edge_dst, float* __restrict__ deg, int E) {
  int i = blockIdx.x * blockDim.x + threadIdx.x;
  if (i < E) atomicAdd(&deg[edge_dst[i]], 1.0f);
}
__global__ void invdeg_kernel(float* __restrict__ deg, int N) {
  int i = blockIdx.x * blockDim.x + threadIdx.x;
  if (i < N) deg[i] = 1.0f / fmaxf(deg[i], 1.0f);
}

// =====================================================================
// gconv: agg[dst] += h[src] @ W[e]  (bf16 W, streamed at HBM rate).
// Block = 16 edges x 16 lanes. The 8KB W tile arrives via a single
// TENSOR_LOAD_TO_LDS DMA (TDM, TENSORcnt) when available, overlapping the
// h-gather; fallback is coalesced b128 loads -> LDS. Sorted dst ->
// block-level segmented reduction: one atomic per dst-run.
// =====================================================================
__global__ void gconv_kernel(const float* __restrict__ h,
                             const __hip_bfloat16* __restrict__ Wb,
                             const int* __restrict__ idxn,
                             const int* __restrict__ edge_dst,
                             float* __restrict__ agg,
                             int E, int N) {
  __shared__ __align__(16) unsigned short Ws[16][16][16]; // [group][row c][col n]
  __shared__ float accs[16][16];
  __shared__ int   dsts[16];

  const int g  = threadIdx.x >> 4;
  const int n  = threadIdx.x & 15;
  const int e  = blockIdx.x * 16 + g;
  const int eC = min(e, E - 1);
  const bool valid = (e < E);

#if USE_TDM
  // --- one 8KB contiguous DMA per block; wave 0 issues, descriptor is uniform ---
  if (threadIdx.x < 32) {
    unsigned long long ga =
        (unsigned long long)(const void*)((const unsigned short*)Wb +
                                          (unsigned long long)blockIdx.x * 16 * 256);
    unsigned lds = (unsigned)(unsigned long long)(void*)&Ws[0][0][0];
    unsigned long long totalQW = (unsigned long long)E * 64;     // W size in 8B units
    unsigned long long baseQW  = (unsigned long long)blockIdx.x * 1024;
    unsigned long long remU    = totalQW - baseQW;
    unsigned remQW = (remU > 0x7FFFFFFFull) ? 0x7FFFFFFFu : (unsigned)remU;

    u32x4 g0;
    g0[0] = 1u;                                           // count=1 (user D#)
    g0[1] = lds;                                          // lds_addr
    g0[2] = (unsigned)(ga & 0xFFFFFFFFull);               // global_addr[31:0]
    g0[3] = (unsigned)((ga >> 32) & 0x01FFFFFFull) | (2u << 30); // addr[56:32] | type=2
    i32x8 g1;
    g1[0] = (int)(3u << 16);                              // data_size=8B, mask=0
    g1[1] = (int)((remQW & 0xFFFFu) << 16);               // tensor_dim0[15:0]
    g1[2] = (int)((remQW >> 16) & 0xFFFFu) | (1 << 16);   // dim0[31:16] | tensor_dim1=1
    g1[3] = (int)(1024u << 16);                           // tile_dim0 = 1024
    g1[4] = 0;                                            // tile_dim1/2 unused
    g1[5] = (int)remQW;                                   // tensor_dim0_stride lo
    g1[6] = 0;
    g1[7] = 0;
    i32x4 gz = {0, 0, 0, 0};
#if __clang_major__ >= 23
    i32x8 gz8 = {0, 0, 0, 0, 0, 0, 0, 0};
    __builtin_amdgcn_tensor_load_to_lds(g0, g1, gz, gz, gz8, 0);
#else
    __builtin_amdgcn_tensor_load_to_lds(g0, g1, gz, gz, 0);
#endif
  }
#else
  // --- fallback: lane n loads W row n (32B) of its edge, coalesced ---
  {
    const uint4* wp = (const uint4*)((const unsigned short*)Wb + (size_t)eC * 256 + n * 16);
    uint4 r0 = wp[0];
    uint4 r1 = wp[1];
    if (e + 16 < E)
      __builtin_prefetch((const unsigned short*)Wb + (size_t)(eC + 16) * 256 + n * 16, 0, 1);
    ((uint4*)&Ws[g][n][0])[0] = r0;
    ((uint4*)&Ws[g][n][0])[1] = r1;
  }
#endif

  // gather h row (overlaps the W-tile transfer)
  const int src = idxn[eC];
  const float4* hp = (const float4*)(h + (size_t)src * NC);
  float4 h0 = hp[0], h1 = hp[1], h2 = hp[2], h3 = hp[3];
  float hv[16] = {h0.x, h0.y, h0.z, h0.w, h1.x, h1.y, h1.z, h1.w,
                  h2.x, h2.y, h2.z, h2.w, h3.x, h3.y, h3.z, h3.w};
  if (n == 0) dsts[g] = edge_dst[eC];

#if USE_TDM
  if (threadIdx.x < 32) __builtin_amdgcn_s_wait_tensorcnt(0);
#endif
  __syncthreads();

  float acc = 0.0f;
#pragma unroll
  for (int c = 0; c < 16; ++c)
    acc += hv[c] * bf16bits_to_f32(Ws[g][c][n]);
  accs[g][n] = acc;
  __syncthreads();

  const int dst = dsts[g];
  const bool head = valid && (g == 0 || dsts[g - 1] != dst);
  if (head) {
    float sum = accs[g][n];
    for (int gg = g + 1; gg < 16 && (blockIdx.x * 16 + gg) < E && dsts[gg] == dst; ++gg)
      sum += accs[gg][n];
    atomicAdd(&agg[(size_t)dst * NC + n], sum);
  }
}

// =====================================================================
// GRU cell via WMMA (wave = 16 nodes). A 16x32 f16: K0..15 = x = agg*inv_deg,
// K16..31 = h. B rows 0..15 = w_ih^T, 16..31 = w_hh^T, staged N-major in LDS
// so each gate-tile B is 2 x ds_load_b128. 4 WMMAs per tile.
// =====================================================================
__global__ void gru_wmma_kernel(const float* __restrict__ agg,
                                const float* __restrict__ invdeg,
                                const float* __restrict__ h_in,
                                const float* __restrict__ w_ih,   // [48,16]
                                const float* __restrict__ w_hh,   // [48,16]
                                const float* __restrict__ b_ih,   // [48]
                                const float* __restrict__ b_hh,   // [48]
                                float* __restrict__ h_out,
                                int N) {
  __shared__ __align__(16) _Float16 wc[48 * 32];   // [col][kr] kr<16: w_ih else w_hh

  const int tid = threadIdx.x;
  for (int i = tid; i < 48 * 32; i += 256) {
    int col = i >> 5, kr = i & 31;
    wc[i] = (_Float16)((kr < NC) ? w_ih[col * NC + kr] : w_hh[col * NC + (kr - NC)]);
  }
  __syncthreads();

  const int wave = tid >> 5;
  const int lane = tid & 31;
  const int hi   = lane >> 4;
  const int mn   = lane & 15;

  const int nTiles = (N + 15) / 16;
  // clamp instead of early-return: duplicate tiles recompute identical values
  const int tile = min(blockIdx.x * (int)(blockDim.x >> 5) + wave, nTiles - 1);
  const int base = tile * 16;

  // A operands: x channels hi*8..hi*8+7 (elems 0..7), h channels (elems 8..15)
  const int nd = min(base + mn, N - 1);
  const float id = invdeg[nd];
  const float4* ap = (const float4*)(agg  + (size_t)nd * NC + hi * 8);
  const float4* hp = (const float4*)(h_in + (size_t)nd * NC + hi * 8);
  float4 x0 = ap[0], x1 = ap[1];
  float4 g0 = hp[0], g1 = hp[1];
  float xs[8] = {x0.x * id, x0.y * id, x0.z * id, x0.w * id,
                 x1.x * id, x1.y * id, x1.z * id, x1.w * id};
  float hs[8] = {g0.x, g0.y, g0.z, g0.w, g1.x, g1.y, g1.z, g1.w};
  v16h ax, ah, axh;
#pragma unroll
  for (int i = 0; i < 8; ++i) {
    _Float16 xv = (_Float16)xs[i];
    _Float16 hv = (_Float16)hs[i];
    ax[i] = xv;              ax[8 + i] = (_Float16)0.0f;
    ah[i] = (_Float16)0.0f;  ah[8 + i] = hv;
    axh[i] = xv;             axh[8 + i] = hv;
  }

  // B operands (2 x ds_load_b128 each)
  v16h bj[3];
#pragma unroll
  for (int j = 0; j < 3; ++j) {
    const _Float16* br = &wc[(j * 16 + mn) * 32];
    bj[j] = cat8(*(const v8h*)(br + hi * 8), *(const v8h*)(br + 16 + hi * 8));
  }

  v8f cr, cz, cnx, cnh;
  {
    float br_ = b_ih[mn]      + b_hh[mn];
    float bz_ = b_ih[16 + mn] + b_hh[16 + mn];
    float bx_ = b_ih[32 + mn];
    float bh_ = b_hh[32 + mn];
#pragma unroll
    for (int v = 0; v < 8; ++v) { cr[v] = br_; cz[v] = bz_; cnx[v] = bx_; cnh[v] = bh_; }
  }
  cr  = __builtin_amdgcn_wmma_f32_16x16x32_f16(false, axh, false, bj[0], (short)0, cr,  false, false);
  cz  = __builtin_amdgcn_wmma_f32_16x16x32_f16(false, axh, false, bj[1], (short)0, cz,  false, false);
  cnx = __builtin_amdgcn_wmma_f32_16x16x32_f16(false, ax,  false, bj[2], (short)0, cnx, false, false);
  cnh = __builtin_amdgcn_wmma_f32_16x16x32_f16(false, ah,  false, bj[2], (short)0, cnh, false, false);

#pragma unroll
  for (int v = 0; v < 8; ++v) {
    int nd2 = base + v + hi * 8;
    float hold = h_in[(size_t)min(nd2, N - 1) * NC + mn];
    float r = sigmoidf_(cr[v]);
    float z = sigmoidf_(cz[v]);
    float nn = tanhf(cnx[v] + r * cnh[v]);
    float hnew = (1.0f - z) * nn + z * hold;
    if (nd2 < N) h_out[(size_t)nd2 * NC + mn] = hnew;
  }
}

// =====================================================================
// Misc: skip-connection add, output pack
// =====================================================================
__global__ void add_kernel(const float* __restrict__ a, const float* __restrict__ b,
                           float* __restrict__ s, int n) {
  int i = blockIdx.x * blockDim.x + threadIdx.x;
  if (i < n) s[i] = a[i] + b[i];
}

__global__ void pack_kernel(const float* __restrict__ s0, const float* __restrict__ s1,
                            const float* __restrict__ s2, const float* __restrict__ s3,
                            const float* __restrict__ s4, const float* __restrict__ s5,
                            const float* __restrict__ s6, const float* __restrict__ s7,
                            const float* __restrict__ s8, const float* __restrict__ s9,
                            const float* __restrict__ s10,
                            float* __restrict__ out, int N) {
  int i = blockIdx.x * blockDim.x + threadIdx.x;
  if (i >= N * NC) return;
  int node = i / NC, c = i % NC;
  const float* srcs[11] = {s0, s1, s2, s3, s4, s5, s6, s7, s8, s9, s10};
  size_t ob = (size_t)node * (11 * NC) + c;
#pragma unroll
  for (int s = 0; s < 11; ++s) out[ob + s * NC] = srcs[s][i];
}

// =====================================================================
// Host orchestration
// =====================================================================
extern "C" void kernel_launch(void* const* d_in, const int* in_sizes, int n_in,
                              void* d_out, int out_size, void* d_ws, size_t ws_size,
                              hipStream_t stream) {
  (void)n_in; (void)out_size; (void)ws_size;
  const float* hx       = (const float*)d_in[0];
  const float* ef       = (const float*)d_in[1];
  const int*   idxn     = (const int*)d_in[2];
  const int*   edge_dst = (const int*)d_in[3];
  const float* w1       = (const float*)d_in[4];
  const float* b1       = (const float*)d_in[5];
  const float* w2       = (const float*)d_in[6];
  const float* b2       = (const float*)d_in[7];
  const float* w_ih     = (const float*)d_in[8];
  const float* w_hh     = (const float*)d_in[9];
  const float* b_ih     = (const float*)d_in[10];
  const float* b_hh     = (const float*)d_in[11];
  float* out = (float*)d_out;

  const int N = in_sizes[0] / NC;
  const int E = in_sizes[1] / FDIM;

  char* ws = (char*)d_ws;
  size_t off = 0;
  auto alignup = [](size_t x) { return (x + 255) & ~(size_t)255; };
  __hip_bfloat16* Wb = (__hip_bfloat16*)(ws + off); off += alignup((size_t)E * 256 * 2);
  float* invdeg = (float*)(ws + off); off += alignup((size_t)N * 4);
  float* agg    = (float*)(ws + off); off += alignup((size_t)N * NC * 4);
  float* st[14];
  for (int i = 0; i < 14; ++i) { st[i] = (float*)(ws + off); off += alignup((size_t)N * NC * 4); }
  // st: 0=h1 1=h2 2=h3 3=s1 4=h4 5=h5 6=s2 7=h6 8=h7 9=s3 10=h8 11=h9 12=s4 13=h10

  fnet_wmma_kernel<<<(E + 127) / 128, 256, 0, stream>>>(ef, w1, b1, w2, b2, Wb, E);

  hipMemsetAsync(invdeg, 0, (size_t)N * 4, stream);
  deg_kernel<<<(E + 255) / 256, 256, 0, stream>>>(edge_dst, invdeg, E);
  invdeg_kernel<<<(N + 255) / 256, 256, 0, stream>>>(invdeg, N);

  const int nTiles = (N + 15) / 16;
  const int gruBlocks = (nTiles + 7) / 8;
  auto cell = [&](const float* hin, float* hout) {
    hipMemsetAsync(agg, 0, (size_t)N * NC * 4, stream);
    gconv_kernel<<<(E + 15) / 16, 256, 0, stream>>>(hin, Wb, idxn, edge_dst, agg, E, N);
    gru_wmma_kernel<<<gruBlocks, 256, 0, stream>>>(agg, invdeg, hin, w_ih, w_hh, b_ih, b_hh, hout, N);
  };
  const int nAdd = (N * NC + 255) / 256;

  cell(hx, st[0]);            // h1
  cell(st[0], st[1]);         // h2
  cell(st[1], st[2]);         // h3
  add_kernel<<<nAdd, 256, 0, stream>>>(st[0], st[2], st[3], N * NC);   // s1 = h1+h3
  cell(st[3], st[4]);         // h4
  cell(st[4], st[5]);         // h5
  add_kernel<<<nAdd, 256, 0, stream>>>(st[2], st[5], st[6], N * NC);   // s2 = h3+h5
  cell(st[6], st[7]);         // h6
  cell(st[7], st[8]);         // h7
  add_kernel<<<nAdd, 256, 0, stream>>>(st[5], st[8], st[9], N * NC);   // s3 = h5+h7
  cell(st[9], st[10]);        // h8
  cell(st[10], st[11]);       // h9
  add_kernel<<<nAdd, 256, 0, stream>>>(st[8], st[11], st[12], N * NC); // s4 = h7+h9
  cell(st[12], st[13]);       // h10

  pack_kernel<<<(N * NC + 255) / 256, 256, 0, stream>>>(
      hx, st[0], st[1], st[3], st[4], st[6], st[7], st[9], st[10], st[12], st[13],
      out, N);
}
